// RNN_53042846105958
// MI455X (gfx1250) — matrix operand order repaired and checked by reading, not compiled, not run on hardware
//
#include <hip/hip_runtime.h>
#include <math.h>

// ---------------------------------------------------------------------------
// Types for WMMA fragments (gfx1250, wave32)
// ---------------------------------------------------------------------------
typedef __attribute__((ext_vector_type(16))) __bf16        bf16x16;
typedef __attribute__((ext_vector_type(8)))  float         f32x8;
typedef __attribute__((ext_vector_type(4)))  unsigned int  u32x4;

union Frag16 {            // 16 bf16 = 8 VGPRs, loaded as two 16B chunks
    u32x4   q[2];
    bf16x16 v;
};

__device__ __forceinline__ unsigned short f2bf(float f) {
    unsigned int u = __builtin_bit_cast(unsigned int, f);
    unsigned int r = u + 0x7FFFu + ((u >> 16) & 1u);   // round-to-nearest-even
    return (unsigned short)(r >> 16);
}

// ---------------------------------------------------------------------------
// fp32 -> bf16 elementwise convert
// ---------------------------------------------------------------------------
__global__ void cvt_f32_bf16_kernel(const float* __restrict__ in,
                                    unsigned short* __restrict__ out,
                                    size_t n) {
    size_t i = (size_t)blockIdx.x * blockDim.x + threadIdx.x;
    if (i < n) out[i] = f2bf(in[i]);
}

// ---------------------------------------------------------------------------
// Transpose [rows][cols] fp32 -> [cols][rows] bf16 (LDS tiled)
// ---------------------------------------------------------------------------
__global__ __launch_bounds__(256)
void transpose_f32_bf16_kernel(const float* __restrict__ w,
                               unsigned short* __restrict__ wT,
                               int rows, int cols) {
    __shared__ float tile[32][33];
    const int tx = threadIdx.x;            // 32
    const int ty = threadIdx.y;            // 8
    const int c0 = blockIdx.x * 32;
    const int r0 = blockIdx.y * 32;
#pragma unroll
    for (int i = ty; i < 32; i += 8)
        tile[i][tx] = w[(size_t)(r0 + i) * cols + (c0 + tx)];
    __syncthreads();
#pragma unroll
    for (int i = ty; i < 32; i += 8)
        wT[(size_t)(c0 + i) * rows + (r0 + tx)] = f2bf(tile[tx][i]);
}

// ---------------------------------------------------------------------------
// WMMA bf16 GEMM:  C[M,N](f32) = (A[M,K] @ BT[N,K]^T + bias) * alpha
// One wave computes a 64x64 tile (4x4 fragments of 16x16), K step = 32.
// A-fragment layout (16-bit A 16x32): lane l holds row (l&15);
//   half = l>>4 selects K base offset {0,8}; VGPR0-3 = K[base..base+7],
//   VGPR4-7 = K[base+16..base+23]  -> two contiguous 16B loads.
// B-fragment layout (16-bit B 32x16): lane l holds col (l&15);
//   half selects K base {0,8}; VGPR0-3 = K[base..base+7], VGPR4-7 = K[base+16..base+23].
// C/D layout: lane l holds col (l&15); VGPR r = row r + 8*(l>>4).
// Operands stay L2-resident (192MB L2), so direct global fragment loads are
// fine: 8KB of cache traffic per 262KFLOP of WMMA per wave per K-step.
// ---------------------------------------------------------------------------
__global__ __launch_bounds__(128)
void wmma_gemm_bf16_kernel(const unsigned short* __restrict__ A,
                           const unsigned short* __restrict__ BT,
                           float* __restrict__ C,
                           const float* __restrict__ bias,
                           int M, int N, int K, float alpha) {
    const int lane = threadIdx.x & 31;
    const int wave = threadIdx.x >> 5;
    const int n0   = blockIdx.x * 64;
    const int m0   = (blockIdx.y * 4 + wave) * 64;
    const int l16  = lane & 15;
    const int half = lane >> 4;

    f32x8 acc[4][4];
#pragma unroll
    for (int i = 0; i < 4; ++i)
#pragma unroll
        for (int j = 0; j < 4; ++j)
#pragma unroll
            for (int r = 0; r < 8; ++r) acc[i][j][r] = 0.0f;

    const unsigned short* aBase = A  + (size_t)(m0 + l16) * K + half * 8;
    const unsigned short* bBase = BT + (size_t)(n0 + l16) * K + half * 8;

    for (int k0 = 0; k0 < K; k0 += 32) {
        Frag16 a[4], b[4];
#pragma unroll
        for (int i = 0; i < 4; ++i) {
            const unsigned short* p = aBase + (size_t)i * 16 * K + k0;
            a[i].q[0] = *(const u32x4*)(p);
            a[i].q[1] = *(const u32x4*)(p + 16);
        }
#pragma unroll
        for (int j = 0; j < 4; ++j) {
            const unsigned short* p = bBase + (size_t)j * 16 * K + k0;
            b[j].q[0] = *(const u32x4*)(p);
            b[j].q[1] = *(const u32x4*)(p + 16);
        }
        if (k0 + 32 < K) {   // near-cache prefetch of next K slab
            __builtin_prefetch(aBase + k0 + 32, 0, 3);
            __builtin_prefetch(bBase + k0 + 32, 0, 3);
        }
#pragma unroll
        for (int i = 0; i < 4; ++i)
#pragma unroll
            for (int j = 0; j < 4; ++j)
                acc[i][j] = __builtin_amdgcn_wmma_f32_16x16x32_bf16(
                    false, a[i].v, false, b[j].v,
                    (short)0, acc[i][j], false, false);
    }

#pragma unroll
    for (int i = 0; i < 4; ++i) {
#pragma unroll
        for (int j = 0; j < 4; ++j) {
            const int col = n0 + j * 16 + l16;
            const float bv = bias ? bias[col] : 0.0f;
#pragma unroll
            for (int r = 0; r < 8; ++r) {
                const int row = m0 + i * 16 + r + half * 8;
                C[(size_t)row * N + col] = (acc[i][j][r] + bv) * alpha;
            }
        }
    }
}

// ---------------------------------------------------------------------------
// Sequential recurrence: one block per (b,h) chain, 64 threads (2 waves).
//   h_new[e] = tanh(u_t[e] + sum_d h[d] * (FACTOR*sw[h][d][e]))
// Weight column in 64 VGPRs/lane; h double-buffered in LDS (1 barrier/step).
// u loads are software-pipelined PF steps deep (register circular buffer;
// in-order load returns -> s_wait_loadcnt <= PF-1) so HBM latency is off the
// per-step critical path. Time loop unrolled xPF so indices/parities are
// compile-time constants. Emits hs rows (bf16) of [B*S, 2048] for GEMM2.
// ---------------------------------------------------------------------------
#define SCAN_PF 8

__global__ __launch_bounds__(64)
void scan_kernel(const float* __restrict__ u,
                 const float* __restrict__ sw,
                 unsigned short* __restrict__ hs,
                 float factor) {
    const int bh = blockIdx.x;          // 0..127
    const int b  = bh >> 5;
    const int h  = bh & 31;
    const int e  = threadIdx.x;         // 0..63

    float wcol[64];
#pragma unroll
    for (int d = 0; d < 64; ++d)
        wcol[d] = sw[((size_t)h * 64 + d) * 64 + e] * factor;

    __shared__ float hbuf[2][64];
    hbuf[0][e] = 0.0f;
    hbuf[1][e] = 0.0f;
    __syncthreads();

    const size_t stride = 2048;
    const float*    up = u  + ((size_t)b * 4096) * stride + h * 64 + e;
    unsigned short* op = hs + ((size_t)b * 4096) * stride + h * 64 + e;

    float ubuf[SCAN_PF];
#pragma unroll
    for (int i = 0; i < SCAN_PF; ++i)
        ubuf[i] = up[(size_t)i * stride];

    for (int t0 = 0; t0 < 4096; t0 += SCAN_PF) {
#pragma unroll
        for (int s = 0; s < SCAN_PF; ++s) {
            const int t = t0 + s;
            float acc = ubuf[s];
            const float* hc = hbuf[s & 1];          // t parity == s parity
#pragma unroll
            for (int d = 0; d < 64; ++d)
                acc = fmaf(hc[d], wcol[d], acc);
            const float hn = tanhf(acc);
            hbuf[(s + 1) & 1][e] = hn;
            op[(size_t)t * stride] = f2bf(hn);
            if (t + SCAN_PF < 4096)                 // refill pipeline slot
                ubuf[s] = up[(size_t)(t + SCAN_PF) * stride];
            __syncthreads();
        }
    }
}

// ---------------------------------------------------------------------------
// Launch: cvt(x) ; transpose(w_in) ; transpose(w_out) ;
//         GEMM1(u = (x@w_in + b)*F) ; scan ; GEMM2(out = hs@w_out)
// ---------------------------------------------------------------------------
extern "C" void kernel_launch(void* const* d_in, const int* in_sizes, int n_in,
                              void* d_out, int out_size, void* d_ws, size_t ws_size,
                              hipStream_t stream) {
    (void)in_sizes; (void)n_in; (void)out_size; (void)ws_size;

    const float* x     = (const float*)d_in[0];   // [4,4096,2048]
    const float* w_in  = (const float*)d_in[1];   // [2048,2048]
    const float* b_in  = (const float*)d_in[2];   // [2048]
    const float* sw    = (const float*)d_in[3];   // [32,64,64]
    const float* w_out = (const float*)d_in[4];   // [2048,2048]
    float*       out   = (float*)d_out;           // [4,4096,2048]

    const int Bq = 4, S = 4096, Din = 2048, Dst = 2048, Dout = 2048;
    const int M = Bq * S;                                   // 16384
    const float FACTOR = 1.0f / sqrtf((float)(Din + 64));   // 1/sqrt(2112)

    // workspace layout (all 64B-aligned power-of-two sizes)
    char* ws = (char*)d_ws;
    unsigned short* xb     = (unsigned short*)(ws);                 //  64 MB
    unsigned short* w_inT  = (unsigned short*)(ws + 67108864);      //   8 MB
    unsigned short* w_outT = (unsigned short*)(ws + 75497472);      //   8 MB
    float*          u      = (float*)        (ws + 83886080);       // 128 MB
    unsigned short* hsb    = (unsigned short*)(ws + 218103808);     //  64 MB

    const size_t nx = (size_t)M * Din;
    cvt_f32_bf16_kernel<<<(unsigned)((nx + 255) / 256), 256, 0, stream>>>(x, xb, nx);

    transpose_f32_bf16_kernel<<<dim3(Dst / 32, Din / 32), dim3(32, 8), 0, stream>>>(
        w_in, w_inT, Din, Dst);
    transpose_f32_bf16_kernel<<<dim3(Dout / 32, Dst / 32), dim3(32, 8), 0, stream>>>(
        w_out, w_outT, Dst, Dout);

    // u = (x @ w_in + b_in) * FACTOR
    wmma_gemm_bf16_kernel<<<dim3(Dst / 64, M / 256), 128, 0, stream>>>(
        xb, w_inT, u, b_in, M, Dst, Din, FACTOR);

    // per-head sequential recurrence -> hs (bf16)
    scan_kernel<<<128, 64, 0, stream>>>(u, sw, hsb, FACTOR);

    // out = hs @ w_out
    wmma_gemm_bf16_kernel<<<dim3(Dout / 64, M / 256), 128, 0, stream>>>(
        hsb, w_outT, out, nullptr, M, Dout, Dst, 1.0f);
}
